// RelativeMultiHeadAttention_21620865368370
// MI455X (gfx1250) — compile-verified, matrix operand order checked by reading, not run on hardware
//
#include <hip/hip_runtime.h>
#include <hip/hip_bf16.h>

// ---------------------------------------------------------------------------
// Relative Multi-Head Attention (Transformer-XL style) for gfx1250 / MI455X.
// All GEMMs use V_WMMA_F32_16X16X32_F16 (f16 operands, fp32 accumulate).
// K-head / V-head operand tiles are staged into LDS with the Tensor Data
// Mover (tensor_load_to_lds + s_wait_tensorcnt).
// BS=4, T=1024, D_MODEL=1024, H=16, D_K=64.
// ---------------------------------------------------------------------------

typedef __attribute__((ext_vector_type(16))) _Float16     v16h;
typedef __attribute__((ext_vector_type(8)))  _Float16     v8h;
typedef __attribute__((ext_vector_type(8)))  float        v8f;
typedef __attribute__((ext_vector_type(4)))  unsigned int v4u;
typedef __attribute__((ext_vector_type(8)))  int          v8i;
typedef __attribute__((ext_vector_type(4)))  int          v4i;

#define T_SEQ  1024
#define DMODEL 1024
#define NHEAD  16
#define DK     64
#define BSZ    4

// ---- WMMA fragment helpers -------------------------------------------------
// A-matrix 16x32 f16 layout (ISA 7.12.2):
//   lanes 0-15 : row M=lane,    halves = K[kc+0..7],  K[kc+16..23]
//   lanes 16-31: row M=lane-16, halves = K[kc+8..15], K[kc+24..31]
// B fragments use the same layout applied to B^T, so any row-major [n][k]
// buffer loads with the identical pattern.

static __device__ __forceinline__ v16h frag_from_f32(
    const float* __restrict__ base, int ld, int row0, int kc, int lane) {
  int r  = row0 + (lane & 15);
  int ko = kc + ((lane >> 4) << 3);
  const float* p = base + (size_t)r * ld + ko;
  v16h f;
#pragma unroll
  for (int i = 0; i < 8; ++i) f[i] = (_Float16)p[i];
#pragma unroll
  for (int i = 0; i < 8; ++i) f[8 + i] = (_Float16)p[16 + i];
  return f;
}

static __device__ __forceinline__ v16h frag_from_f16(
    const _Float16* __restrict__ base, int ld, int row0, int kc, int lane) {
  int r  = row0 + (lane & 15);
  int ko = kc + ((lane >> 4) << 3);
  const _Float16* p = base + (size_t)r * ld + ko;
  v8h lo = *(const v8h*)p;
  v8h hi = *(const v8h*)(p + 16);
  return __builtin_shufflevector(lo, hi, 0, 1, 2, 3, 4, 5, 6, 7,
                                 8, 9, 10, 11, 12, 13, 14, 15);
}

static __device__ __forceinline__ v8f wmma16(v16h a, v16h b, v8f c) {
  return __builtin_amdgcn_wmma_f32_16x16x32_f16(
      /*neg_a=*/false, a, /*neg_b=*/false, b,
      /*c_mod=*/(short)0, c, /*reuse_a=*/false, /*reuse_b=*/false);
}

// ---- Tensor Data Mover: stage a 2-D f16 tile (rows x rowlen, row pitch in
// elements) from global memory into LDS.  D# bit layout per ISA ch.8.
// Issued by wave 0 only (scalar-uniform branch); TDM ignores EXEC.
// This toolchain exposes the 6-arg builtin:
//   (uint32x4 g0, int32x8 g1, int32x4 g2, int32x4 g3, int32x8 extra, i32 cpol)
static __device__ __forceinline__ void tdm_load_2d_f16(
    const _Float16* gsrc, void* lds_dst,
    unsigned rowlen, unsigned rows, unsigned pitch) {
  unsigned long long ga = (unsigned long long)(size_t)gsrc;
  unsigned lds = (unsigned)(size_t)lds_dst;
  // group0: count=1 | lds_addr | global_addr[56:0] | type=2
  v4u g0 = { 1u, lds, (unsigned)ga,
             ((unsigned)(ga >> 32) & 0x01FFFFFFu) | 0x80000000u };
  // group1: data_size=2B; tensor_dim0=rowlen, tensor_dim1=rows,
  //         tile_dim0=rowlen, tile_dim1=rows, tile_dim2=0,
  //         tensor_dim0_stride=pitch (elements), dim1 stride unused (2D).
  v8i g1 = { (int)0x00010000u,            // wg_mask=0, data_size=1 (2 bytes)
             (int)(rowlen << 16),          // barrier_addr=0 | tensor_dim0 lo
             (int)(rows << 16),            // tensor_dim0 hi=0 | tensor_dim1 lo
             (int)(rowlen << 16),          // tensor_dim1 hi=0 | tile_dim0
             (int)rows,                    // tile_dim1 | tile_dim2=0
             (int)pitch,                   // tensor_dim0_stride[31:0]
             0, 0 };
  v4i z4 = { 0, 0, 0, 0 };
  v8i z8 = { 0, 0, 0, 0, 0, 0, 0, 0 };
  __builtin_amdgcn_tensor_load_to_lds(g0, g1, z4, z4, z8, 0);
}

// ---------------------------------------------------------------------------
// K1: projection GEMM.  C[M,N] = A[M,1024] @ W[N,1024]^T (+bias).
// MODE 0: two f16 outputs (qh + pos_bias_u, qh + pos_bias_v)
// MODE 1: one f16 output, +bias                 (K projection)
// MODE 2: one f16 output transposed [N][4096], +bias  (V projection)
// MODE 3: one f16 output, no bias               (P projection)
// Grid: (N/64, M/64), block 256 (8 waves), tile 64x64, wave = 16x32 sub-tile.
// ---------------------------------------------------------------------------
template <int MODE>
__global__ __launch_bounds__(256) void proj_gemm_f32(
    const float* __restrict__ A, const float* __restrict__ W,
    const float* __restrict__ bias,
    _Float16* __restrict__ out0, const float* __restrict__ add0,
    _Float16* __restrict__ out1, const float* __restrict__ add1) {
  int lane = threadIdx.x & 31;
  int w    = threadIdx.x >> 5;
  int row0 = blockIdx.y * 64 + 16 * (w & 3);
  int col0 = blockIdx.x * 64 + 32 * (w >> 2);

  v8f acc0 = {}, acc1 = {};
  for (int kc = 0; kc < DMODEL; kc += 32) {
    v16h a  = frag_from_f32(A, DMODEL, row0, kc, lane);
    v16h b0 = frag_from_f32(W, DMODEL, col0, kc, lane);
    v16h b1 = frag_from_f32(W, DMODEL, col0 + 16, kc, lane);
    acc0 = wmma16(a, b0, acc0);
    acc1 = wmma16(a, b1, acc1);
  }

  int cl   = lane & 15;
  int rOff = (lane >> 4) << 3;
#pragma unroll
  for (int t = 0; t < 2; ++t) {
    v8f acc = t ? acc1 : acc0;
    int c = col0 + 16 * t + cl;
    float bv = (MODE == 3) ? 0.0f : bias[c];
#pragma unroll
    for (int vg = 0; vg < 8; ++vg) {
      int r = row0 + rOff + vg;
      float v = acc[vg] + bv;
      if (MODE == 0) {
        out0[(size_t)r * DMODEL + c] = (_Float16)(v + add0[c]);
        out1[(size_t)r * DMODEL + c] = (_Float16)(v + add1[c]);
      } else if (MODE == 2) {
        out0[(size_t)c * (BSZ * T_SEQ) + r] = (_Float16)v;
      } else {
        out0[(size_t)r * DMODEL + c] = (_Float16)v;
      }
    }
  }
}

// ---------------------------------------------------------------------------
// K2: BD (unshifted) = q_v @ P^T per (b,h).  M=N=1024, K=64.
// Grid: (16, 16, BS*H).
// ---------------------------------------------------------------------------
__global__ __launch_bounds__(256) void bd_gemm(
    const _Float16* __restrict__ QV, const _Float16* __restrict__ P16,
    float* __restrict__ BD) {
  int lane = threadIdx.x & 31;
  int w    = threadIdx.x >> 5;
  int bh   = blockIdx.z;
  int b = bh >> 4, h = bh & 15;
  const _Float16* Aq = QV + (size_t)b * T_SEQ * DMODEL + h * DK;
  const _Float16* Bp = P16 + h * DK;
  float* out = BD + (size_t)bh * T_SEQ * T_SEQ;

  int row0 = blockIdx.y * 64 + 16 * (w & 3);
  int col0 = blockIdx.x * 64 + 32 * (w >> 2);

  v8f acc0 = {}, acc1 = {};
#pragma unroll
  for (int kc = 0; kc < DK; kc += 32) {
    v16h a  = frag_from_f16(Aq, DMODEL, row0, kc, lane);
    v16h b0 = frag_from_f16(Bp, DMODEL, col0, kc, lane);
    v16h b1 = frag_from_f16(Bp, DMODEL, col0 + 16, kc, lane);
    acc0 = wmma16(a, b0, acc0);
    acc1 = wmma16(a, b1, acc1);
  }

  int cl = lane & 15, rOff = (lane >> 4) << 3;
#pragma unroll
  for (int vg = 0; vg < 8; ++vg) {
    int r = row0 + rOff + vg;
    out[(size_t)r * T_SEQ + col0 + cl]      = acc0[vg];
    out[(size_t)r * T_SEQ + col0 + 16 + cl] = acc1[vg];
  }
}

// ---------------------------------------------------------------------------
// K3: scores = (q_u @ K^T + rel_shift(BD)) / 8, mask, softmax -> attn (fp32).
// The whole K-head slice (1024 x 64 f16 = 128 KB) is staged into LDS by the
// TDM; all 8 waves read B fragments from LDS.
// rel_shift gather (pad+reshape with pos_len == T):
//   j <= i   : BD[i, T-1-i+j]
//   j == i+1 : 0
//   j >  i+1 : BD[i+1, j-i-2]
// Grid: (T/32, BS*H).  Workgroup: 32 rows x 1024 cols, 8 waves.
// ---------------------------------------------------------------------------
__global__ __launch_bounds__(256) void attn_softmax(
    const _Float16* __restrict__ QU, const _Float16* __restrict__ KH,
    const float* __restrict__ BD, const int* __restrict__ mask,
    float* __restrict__ attn) {
  __shared__ _Float16 ksm[T_SEQ * DK];   // [s][d], ld = 64 (128 KB)
  __shared__ float redmax[4][32];
  __shared__ float redsum[4][32];

  int lane = threadIdx.x & 31;
  int w    = threadIdx.x >> 5;
  int bh   = blockIdx.y;
  int b = bh >> 4, h = bh & 15;
  int t0    = blockIdx.x * 32;
  int row16 = t0 + 16 * (w & 1);
  int cw    = w >> 1;           // column-wave 0..3
  int colbase = cw * 256;

  const _Float16* Aq = QU + (size_t)b * T_SEQ * DMODEL + h * DK;
  const _Float16* Bk = KH + (size_t)b * T_SEQ * DMODEL + h * DK;
  const float*    bd = BD + (size_t)bh * T_SEQ * T_SEQ;
  const int*    mrow = mask + (size_t)b * T_SEQ * T_SEQ;
  float*        arow = attn + (size_t)bh * T_SEQ * T_SEQ;

  // --- TDM stage: K-head slice -> LDS (wave 0 issues, all waves consume) ---
  if (__builtin_amdgcn_readfirstlane(w) == 0) {
    tdm_load_2d_f16(Bk, ksm, DK, T_SEQ, DMODEL);
    __builtin_amdgcn_s_wait_tensorcnt(0);
  }
  __syncthreads();

  v16h a0 = frag_from_f16(Aq, DMODEL, row16, 0, lane);
  v16h a1 = frag_from_f16(Aq, DMODEL, row16, 32, lane);

  v8f acc[16];
#pragma unroll
  for (int ct = 0; ct < 16; ++ct) {
    int c0 = colbase + ct * 16;
    v16h b0 = frag_from_f16(ksm, DK, c0, 0, lane);
    v16h b1 = frag_from_f16(ksm, DK, c0, 32, lane);
    v8f z = {};
    z = wmma16(a0, b0, z);
    z = wmma16(a1, b1, z);
    acc[ct] = z;
  }

  int cl = lane & 15, rOff = (lane >> 4) << 3;
  int lw = 16 * (w & 1);

  float mx[8];
#pragma unroll
  for (int vg = 0; vg < 8; ++vg) mx[vg] = -3.0e38f;

#pragma unroll
  for (int ct = 0; ct < 16; ++ct) {
    int c = colbase + ct * 16 + cl;
#pragma unroll
    for (int vg = 0; vg < 8; ++vg) {
      int r = row16 + rOff + vg;
      float bdv;
      if (c <= r)            bdv = bd[(size_t)r * T_SEQ + (T_SEQ - 1 - r + c)];
      else if (c == r + 1)   bdv = 0.0f;
      else                   bdv = bd[(size_t)(r + 1) * T_SEQ + (c - r - 2)];
      float s = (acc[ct][vg] + bdv) * 0.125f;  // 1/sqrt(64)
      if (mrow[(size_t)r * T_SEQ + c] == 0) s = -10000.0f;
      acc[ct][vg] = s;
      mx[vg] = fmaxf(mx[vg], s);
    }
  }

  // row-max: reduce across 16 lanes sharing the same rows, then across waves
#pragma unroll
  for (int vg = 0; vg < 8; ++vg)
#pragma unroll
    for (int off = 1; off < 16; off <<= 1)
      mx[vg] = fmaxf(mx[vg], __shfl_xor(mx[vg], off, 32));
  if (cl == 0) {
#pragma unroll
    for (int vg = 0; vg < 8; ++vg) redmax[cw][lw + rOff + vg] = mx[vg];
  }
  __syncthreads();

  float sum[8];
#pragma unroll
  for (int vg = 0; vg < 8; ++vg) {
    int lr = lw + rOff + vg;
    mx[vg] = fmaxf(fmaxf(redmax[0][lr], redmax[1][lr]),
                   fmaxf(redmax[2][lr], redmax[3][lr]));
    sum[vg] = 0.0f;
  }

#pragma unroll
  for (int ct = 0; ct < 16; ++ct)
#pragma unroll
    for (int vg = 0; vg < 8; ++vg) {
      float p = __expf(acc[ct][vg] - mx[vg]);
      acc[ct][vg] = p;
      sum[vg] += p;
    }

#pragma unroll
  for (int vg = 0; vg < 8; ++vg)
#pragma unroll
    for (int off = 1; off < 16; off <<= 1)
      sum[vg] += __shfl_xor(sum[vg], off, 32);
  if (cl == 0) {
#pragma unroll
    for (int vg = 0; vg < 8; ++vg) redsum[cw][lw + rOff + vg] = sum[vg];
  }
  __syncthreads();

#pragma unroll
  for (int vg = 0; vg < 8; ++vg) {
    int lr = lw + rOff + vg;
    float s = redsum[0][lr] + redsum[1][lr] + redsum[2][lr] + redsum[3][lr];
    sum[vg] = 1.0f / s;
  }

#pragma unroll
  for (int ct = 0; ct < 16; ++ct) {
    int c = colbase + ct * 16 + cl;
#pragma unroll
    for (int vg = 0; vg < 8; ++vg) {
      int r = row16 + rOff + vg;
      arow[(size_t)r * T_SEQ + c] = acc[ct][vg] * sum[vg];
    }
  }
}

// ---------------------------------------------------------------------------
// K4a: out_h = attn @ V  per (b,h).  M=1024 (t), N=64 (d), K=1024 (s).
// The V-head slice (64 x 1024 f16 = 128 KB, transposed layout [d][b*T+s])
// is TDM-staged into LDS once per workgroup.
// Writes concat layout CC[b*T+t][h*64+d] (f16).  Grid: (T/64, BS*H).
// ---------------------------------------------------------------------------
__global__ __launch_bounds__(256) void av_gemm(
    const float* __restrict__ attn, const _Float16* __restrict__ VHT,
    _Float16* __restrict__ CC) {
  __shared__ _Float16 vsm[DK * T_SEQ];   // [d][s], ld = 1024 (128 KB)

  int lane = threadIdx.x & 31;
  int w    = threadIdx.x >> 5;
  int bh   = blockIdx.y;
  int b = bh >> 4, h = bh & 15;
  const float*    Aa = attn + (size_t)bh * T_SEQ * T_SEQ;
  const _Float16* Bv = VHT + (size_t)(h * DK) * (BSZ * T_SEQ) + b * T_SEQ;

  if (__builtin_amdgcn_readfirstlane(w) == 0) {
    tdm_load_2d_f16(Bv, vsm, T_SEQ, DK, BSZ * T_SEQ);
    __builtin_amdgcn_s_wait_tensorcnt(0);
  }
  __syncthreads();

  int row0 = blockIdx.x * 64 + 16 * (w & 3);   // t
  int col0 = 32 * (w >> 2);                    // d within head

  v8f acc0 = {}, acc1 = {};
  for (int kc = 0; kc < T_SEQ; kc += 32) {
    v16h a  = frag_from_f32(Aa, T_SEQ, row0, kc, lane);
    v16h b0 = frag_from_f16(vsm, T_SEQ, col0, kc, lane);
    v16h b1 = frag_from_f16(vsm, T_SEQ, col0 + 16, kc, lane);
    acc0 = wmma16(a, b0, acc0);
    acc1 = wmma16(a, b1, acc1);
  }

  int cl = lane & 15, rOff = (lane >> 4) << 3;
#pragma unroll
  for (int vg = 0; vg < 8; ++vg) {
    int r = row0 + rOff + vg;
    _Float16* dst = CC + (size_t)(b * T_SEQ + r) * DMODEL + h * DK;
    dst[col0 + cl]      = (_Float16)acc0[vg];
    dst[col0 + 16 + cl] = (_Float16)acc1[vg];
  }
}

// ---------------------------------------------------------------------------
// K4b: output = concat @ Wo^T + bo.  M=4096, N=K=1024, fp32 out.
// Grid: (16, 64).
// ---------------------------------------------------------------------------
__global__ __launch_bounds__(256) void out_gemm(
    const _Float16* __restrict__ CC, const float* __restrict__ Wo,
    const float* __restrict__ bo, float* __restrict__ out) {
  int lane = threadIdx.x & 31;
  int w    = threadIdx.x >> 5;
  int row0 = blockIdx.y * 64 + 16 * (w & 3);
  int col0 = blockIdx.x * 64 + 32 * (w >> 2);

  v8f acc0 = {}, acc1 = {};
  for (int kc = 0; kc < DMODEL; kc += 32) {
    v16h a  = frag_from_f16(CC, DMODEL, row0, kc, lane);
    v16h b0 = frag_from_f32(Wo, DMODEL, col0, kc, lane);
    v16h b1 = frag_from_f32(Wo, DMODEL, col0 + 16, kc, lane);
    acc0 = wmma16(a, b0, acc0);
    acc1 = wmma16(a, b1, acc1);
  }

  int cl = lane & 15, rOff = (lane >> 4) << 3;
#pragma unroll
  for (int vg = 0; vg < 8; ++vg) {
    int r  = row0 + rOff + vg;
    int c0 = col0 + cl, c1 = col0 + 16 + cl;
    out[(size_t)r * DMODEL + c0] = acc0[vg] + bo[c0];
    out[(size_t)r * DMODEL + c1] = acc1[vg] + bo[c1];
  }
}

// ---------------------------------------------------------------------------
// Host launcher.  Workspace layout (f16 unless noted):
//   QU[4096][1024], QV[4096][1024], KH[4096][1024], VHT[1024][4096],
//   P16[1024][1024], CC[4096][1024], BD fp32 [64][1024][1024].
// Total ~298 MB.
// ---------------------------------------------------------------------------
extern "C" void kernel_launch(void* const* d_in, const int* in_sizes, int n_in,
                              void* d_out, int out_size, void* d_ws,
                              size_t ws_size, hipStream_t stream) {
  (void)in_sizes; (void)n_in; (void)out_size; (void)ws_size;

  const float* q   = (const float*)d_in[0];
  const float* k   = (const float*)d_in[1];
  const float* v   = (const float*)d_in[2];
  const float* pe  = (const float*)d_in[3];
  const int*  mask = (const int*)d_in[4];
  const float* Wq  = (const float*)d_in[5];
  const float* bq  = (const float*)d_in[6];
  const float* Wk  = (const float*)d_in[7];
  const float* bk  = (const float*)d_in[8];
  const float* Wv  = (const float*)d_in[9];
  const float* bv  = (const float*)d_in[10];
  const float* Wp  = (const float*)d_in[11];
  const float* pbu = (const float*)d_in[12];  // (H, DK) flat == 1024 cols
  const float* pbv = (const float*)d_in[13];
  const float* Wo  = (const float*)d_in[14];
  const float* bo  = (const float*)d_in[15];

  float* out  = (float*)d_out;
  float* attn = out + (size_t)BSZ * T_SEQ * DMODEL;

  const size_t NTD = (size_t)BSZ * T_SEQ * DMODEL;  // 4M elements
  _Float16* QU  = (_Float16*)d_ws;
  _Float16* QV  = QU + NTD;
  _Float16* KH  = QV + NTD;
  _Float16* VHT = KH + NTD;
  _Float16* P16 = VHT + NTD;
  _Float16* CC  = P16 + (size_t)T_SEQ * DMODEL;
  float*    BD  = (float*)(CC + NTD);

  dim3 blk(256);
  // projections
  proj_gemm_f32<0><<<dim3(16, 64), blk, 0, stream>>>(q, Wq, bq, QU, pbu, QV, pbv);
  proj_gemm_f32<1><<<dim3(16, 64), blk, 0, stream>>>(k, Wk, bk, KH, nullptr, nullptr, nullptr);
  proj_gemm_f32<2><<<dim3(16, 64), blk, 0, stream>>>(v, Wv, bv, VHT, nullptr, nullptr, nullptr);
  proj_gemm_f32<3><<<dim3(16, 16), blk, 0, stream>>>(pe, Wp, nullptr, P16, nullptr, nullptr, nullptr);
  // positional term (unshifted)
  bd_gemm<<<dim3(16, 16, BSZ * NHEAD), blk, 0, stream>>>(QV, P16, BD);
  // scores + rel_shift + softmax -> attn (fp32 output #2)
  attn_softmax<<<dim3(T_SEQ / 32, BSZ * NHEAD), blk, 0, stream>>>(QU, KH, BD, mask, attn);
  // attn @ V -> concat
  av_gemm<<<dim3(T_SEQ / 64, BSZ * NHEAD), blk, 0, stream>>>(attn, VHT, CC);
  // concat @ Wo^T + bo -> output (fp32 output #1)
  out_gemm<<<dim3(16, 64), blk, 0, stream>>>(CC, Wo, bo, out);
}
